// ClassicalAttentionLayer_59562606461504
// MI455X (gfx1250) — compile-verified, matrix operand order checked by reading, not run on hardware
//
#include <hip/hip_runtime.h>
#include <hip/hip_bf16.h>

// Sizes fixed by the reference.
#define NT 8192   // sequence length N
#define DI 1024   // d_in
#define DO 1024   // d_out

typedef __attribute__((ext_vector_type(16))) _Float16 v16h;
typedef __attribute__((ext_vector_type(8)))  _Float16 v8h;
typedef __attribute__((ext_vector_type(4)))  _Float16 v4h;
typedef __attribute__((ext_vector_type(8)))  float    v8f;
typedef __attribute__((ext_vector_type(4)))  float    v4f;

#define WMMA_F16(A, B, C) \
  __builtin_amdgcn_wmma_f32_16x16x32_f16(false, (A), false, (B), (short)0, (C), false, false)

// ---------------------------------------------------------------------------
// Fragment loader.
// CDNA5 16-bit A (16x32, M in lanes) and B (32x16, N in lanes) layouts both
// reduce to: lane L reads, from row (L&15) of a row-major source, two
// contiguous 8-halfword chunks at k-offsets c and c+16 where c=(L>>4)*8.
// ---------------------------------------------------------------------------
__device__ __forceinline__ v16h load_frag(const _Float16* base, int row0, int ld, int kofs) {
  const int lane = threadIdx.x & 31;
  const _Float16* p = base + (size_t)(row0 + (lane & 15)) * (size_t)ld
                           + (size_t)(kofs + ((lane >> 4) << 3));
  v8h lo = *(const v8h*)p;          // 16B load
  v8h hi = *(const v8h*)(p + 16);   // 16B load
  return __builtin_shufflevector(lo, hi, 0,1,2,3,4,5,6,7,8,9,10,11,12,13,14,15);
}

// ---------------------------------------------------------------------------
// fp32 -> f16 conversion (grid-strided, 4 elements/thread/iter)
// ---------------------------------------------------------------------------
__global__ void cvt_f32_f16(const float* __restrict__ src, _Float16* __restrict__ dst, int n4) {
  int i = blockIdx.x * blockDim.x + threadIdx.x;
  const int stride = gridDim.x * blockDim.x;
  for (; i < n4; i += stride) {
    v4f v = ((const v4f*)src)[i];
    v4h b;
    b[0] = (_Float16)v[0]; b[1] = (_Float16)v[1];
    b[2] = (_Float16)v[2]; b[3] = (_Float16)v[3];
    ((v4h*)dst)[i] = b;
  }
}

// ---------------------------------------------------------------------------
// Projection GEMM: Y = X @ W^T.  X:[NT,DI] f16 row-major, W:[DO,DI] f16
// row-major (B[k][j] = W[j][k] -> lane reads row j of W, contiguous in k).
// Each wave computes a 64x32 output region (4x2 tiles, 8 accumulators).
// transposeOut=0: Y row-major [NT,DO].  transposeOut=1: store Y^T [DO,NT].
// ---------------------------------------------------------------------------
__global__ __launch_bounds__(256)
void proj_f16(const _Float16* __restrict__ X, const _Float16* __restrict__ W,
              _Float16* __restrict__ out, int transposeOut) {
  const int wave = threadIdx.x >> 5;
  const int lane = threadIdx.x & 31;
  const int job  = blockIdx.x * 8 + wave;   // 4096 wave-jobs = 128 rowgrp x 32 colgrp
  const int r0   = (job >> 5) * 64;
  const int c0   = (job & 31) * 32;

  const v8f vzero = {0.f,0.f,0.f,0.f,0.f,0.f,0.f,0.f};
  v8f acc[4][2];
#pragma unroll
  for (int m = 0; m < 4; ++m) { acc[m][0] = vzero; acc[m][1] = vzero; }

  for (int k0 = 0; k0 < DI; k0 += 32) {
    v16h a[4];
#pragma unroll
    for (int m = 0; m < 4; ++m) a[m] = load_frag(X, r0 + m * 16, DI, k0);
#pragma unroll
    for (int n = 0; n < 2; ++n) {
      v16h b = load_frag(W, c0 + n * 16, DI, k0);
#pragma unroll
      for (int m = 0; m < 4; ++m) acc[m][n] = WMMA_F16(a[m], b, acc[m][n]);
    }
  }

  const int h  = lane >> 4;   // C layout: elem e of vgpr <-> row e + 8*h
  const int cc = lane & 15;   // col = lane&15
  if (!transposeOut) {
#pragma unroll
    for (int m = 0; m < 4; ++m)
#pragma unroll
      for (int n = 0; n < 2; ++n)
#pragma unroll
        for (int e = 0; e < 8; ++e)
          out[(size_t)(r0 + m * 16 + e + 8 * h) * DO + (c0 + n * 16 + cc)] =
              (_Float16)acc[m][n][e];
  } else {
    // Lane holds 8 consecutive rows of one column -> one packed 16B store.
#pragma unroll
    for (int m = 0; m < 4; ++m)
#pragma unroll
      for (int n = 0; n < 2; ++n) {
        v8h pk;
#pragma unroll
        for (int e = 0; e < 8; ++e) pk[e] = (_Float16)acc[m][n][e];
        *(v8h*)(out + (size_t)(c0 + n * 16 + cc) * NT + (r0 + m * 16 + 8 * h)) = pk;
      }
  }
}

// ---------------------------------------------------------------------------
// Flash attention: block = 32 query rows (2 q-tiles), 8 waves.
//   scores = (Q @ K^T) / NT, softmax over keys, O = P @ V.
// Wave w owns d-slice [128w,128w+128) for both the QK^T partial reduction
// (combined across waves via ds_add_f32 into LDS) and its 128 output cols.
// K-fragments are shared across both q-tiles (QK^T) and V-fragments across
// both P-tiles (PV): 32 WMMAs per 32 x b128 loads per KV block per wave.
// Online softmax is distributed: wave w owns score rows 4w..4w+3 (2 rows per
// 16-lane half-group, reduced with __shfl_xor width 16).
// ---------------------------------------------------------------------------
__global__ __launch_bounds__(256)
void attn_flash(const _Float16* __restrict__ Q, const _Float16* __restrict__ K,
                const _Float16* __restrict__ Vt, float* __restrict__ out) {
  __shared__ float    sS[32][32];
  __shared__ _Float16 sP[32][32];
  __shared__ float    sScale[32];
  __shared__ float    sLinv[32];

  const int wave = threadIdx.x >> 5;
  const int lane = threadIdx.x & 31;
  const int q0   = blockIdx.x * 32;
  const int wd0  = wave * 128;
  const int h    = lane >> 4;
  const int cc   = lane & 15;
  const float invN = 1.0f / (float)NT;

  for (int i = threadIdx.x; i < 32 * 32; i += 256) (&sS[0][0])[i] = 0.0f;

  const v8f vzero = {0.f,0.f,0.f,0.f,0.f,0.f,0.f,0.f};

  // Persistent Q fragments: 2 q-tiles x 4 k-steps over this wave's d-slice.
  v16h qf[2][4];
#pragma unroll
  for (int qt = 0; qt < 2; ++qt)
#pragma unroll
    for (int ks = 0; ks < 4; ++ks)
      qf[qt][ks] = load_frag(Q, q0 + qt * 16, DI, wd0 + ks * 32);

  // O accumulators: 2 q-tiles x 8 d-tiles (16 x v8f = 128 VGPRs).
  v8f acc[2][8];
#pragma unroll
  for (int qt = 0; qt < 2; ++qt)
#pragma unroll
    for (int t = 0; t < 8; ++t) acc[qt][t] = vzero;

  // Online-softmax state for the 2 rows this half-group owns.
  float m_run[2] = {-INFINITY, -INFINITY};
  float l_run[2] = {0.0f, 0.0f};

  __syncthreads();

  for (int jb = 0; jb < NT / 32; ++jb) {
    const int j0 = jb * 32;

    // Prefetch next KV block (global_prefetch_b8 hint; L2-resident anyway).
    if (jb + 1 < NT / 32) {
      __builtin_prefetch(K + (size_t)(j0 + 32 + cc + 16 * h) * DI + wd0, 0, 1);
      __builtin_prefetch(Vt + (size_t)(wd0 + cc + 16 * h) * NT + (j0 + 32), 0, 1);
    }

    // --- partial S = Q K^T over this wave's 128-wide d slice ---
    v8f s[2][2];
    s[0][0] = vzero; s[0][1] = vzero; s[1][0] = vzero; s[1][1] = vzero;
#pragma unroll
    for (int ks = 0; ks < 4; ++ks) {
      v16h kf0 = load_frag(K, j0,      DI, wd0 + ks * 32);
      v16h kf1 = load_frag(K, j0 + 16, DI, wd0 + ks * 32);
      s[0][0] = WMMA_F16(qf[0][ks], kf0, s[0][0]);
      s[0][1] = WMMA_F16(qf[0][ks], kf1, s[0][1]);
      s[1][0] = WMMA_F16(qf[1][ks], kf0, s[1][0]);
      s[1][1] = WMMA_F16(qf[1][ks], kf1, s[1][1]);
    }
#pragma unroll
    for (int qt = 0; qt < 2; ++qt)
#pragma unroll
      for (int kt = 0; kt < 2; ++kt)
#pragma unroll
        for (int e = 0; e < 8; ++e)
          atomicAdd(&sS[qt * 16 + e + 8 * h][kt * 16 + cc], s[qt][kt][e]);
    __syncthreads();

    // --- distributed online softmax: wave w handles rows 4w..4w+3 ---
#pragma unroll
    for (int sr = 0; sr < 2; ++sr) {
      const int r = wave * 4 + h * 2 + sr;
      float v1 = sS[r][cc]      * invN;
      float v2 = sS[r][cc + 16] * invN;
      sS[r][cc] = 0.0f; sS[r][cc + 16] = 0.0f;   // re-zero for next ds_adds
      float bm = fmaxf(v1, v2);
#pragma unroll
      for (int mM = 8; mM >= 1; mM >>= 1) bm = fmaxf(bm, __shfl_xor(bm, mM, 16));
      const float mn = fmaxf(m_run[sr], bm);
      const float sc = __expf(m_run[sr] - mn);
      const float p1 = __expf(v1 - mn);
      const float p2 = __expf(v2 - mn);
      float sum = p1 + p2;
#pragma unroll
      for (int mM = 8; mM >= 1; mM >>= 1) sum += __shfl_xor(sum, mM, 16);
      m_run[sr] = mn;
      l_run[sr] = l_run[sr] * sc + sum;
      sP[r][cc]      = (_Float16)p1;
      sP[r][cc + 16] = (_Float16)p2;
      if (cc == 0) sScale[r] = sc;
    }
    __syncthreads();

    // --- rescale running O, then O += P @ V over this wave's 128 cols ---
    float scv[2][8];
#pragma unroll
    for (int qt = 0; qt < 2; ++qt)
#pragma unroll
      for (int e = 0; e < 8; ++e) scv[qt][e] = sScale[qt * 16 + e + 8 * h];
#pragma unroll
    for (int qt = 0; qt < 2; ++qt)
#pragma unroll
      for (int t = 0; t < 8; ++t)
#pragma unroll
        for (int e = 0; e < 8; ++e) acc[qt][t][e] *= scv[qt][e];

    v16h pf[2];
#pragma unroll
    for (int qt = 0; qt < 2; ++qt) {
      const _Float16* pp = &sP[qt * 16 + cc][h * 8];
      v8h plo = *(const v8h*)pp;
      v8h phi = *(const v8h*)(pp + 16);
      pf[qt] = __builtin_shufflevector(plo, phi, 0,1,2,3,4,5,6,7,8,9,10,11,12,13,14,15);
    }
#pragma unroll
    for (int t = 0; t < 8; ++t) {
      v16h vf = load_frag(Vt, wd0 + t * 16, NT, j0);   // shared by both q-tiles
      acc[0][t] = WMMA_F16(pf[0], vf, acc[0][t]);
      acc[1][t] = WMMA_F16(pf[1], vf, acc[1][t]);
    }
  }

  if (cc == 0) {
    sLinv[wave * 4 + h * 2 + 0] = 1.0f / l_run[0];
    sLinv[wave * 4 + h * 2 + 1] = 1.0f / l_run[1];
  }
  __syncthreads();

  float li[2][8];
#pragma unroll
  for (int qt = 0; qt < 2; ++qt)
#pragma unroll
    for (int e = 0; e < 8; ++e) li[qt][e] = sLinv[qt * 16 + e + 8 * h];
#pragma unroll
  for (int qt = 0; qt < 2; ++qt)
#pragma unroll
    for (int t = 0; t < 8; ++t)
#pragma unroll
      for (int e = 0; e < 8; ++e)
        out[(size_t)(q0 + qt * 16 + e + 8 * h) * DO + (wd0 + t * 16 + cc)] =
            acc[qt][t][e] * li[qt][e];
}

// ---------------------------------------------------------------------------
// Launch: convert x/Wq/Wk/Wv to f16, run 3 WMMA projections (V transposed),
// then flash attention.  Workspace: 70 MiB.
// ---------------------------------------------------------------------------
extern "C" void kernel_launch(void* const* d_in, const int* in_sizes, int n_in,
                              void* d_out, int out_size, void* d_ws, size_t ws_size,
                              hipStream_t stream) {
  const float* x  = (const float*)d_in[0];
  const float* Wq = (const float*)d_in[1];
  const float* Wk = (const float*)d_in[2];
  const float* Wv = (const float*)d_in[3];
  float* out = (float*)d_out;

  _Float16* xb  = (_Float16*)d_ws;                       // [NT, DI]
  _Float16* wqb = xb  + (size_t)NT * DI;                 // [DO, DI]
  _Float16* wkb = wqb + (size_t)DO * DI;
  _Float16* wvb = wkb + (size_t)DO * DI;
  _Float16* qh  = wvb + (size_t)DO * DI;                 // [NT, DO]
  _Float16* kh  = qh  + (size_t)NT * DO;                 // [NT, DO]
  _Float16* vth = kh  + (size_t)NT * DO;                 // [DO, NT] (transposed)
  // total: (8M + 3*1M + 3*8M) * 2B = 70 MiB <= ws_size

  cvt_f32_f16<<<256, 256, 0, stream>>>(x,  xb,  (NT * DI) / 4);
  cvt_f32_f16<<<64,  256, 0, stream>>>(Wq, wqb, (DO * DI) / 4);
  cvt_f32_f16<<<64,  256, 0, stream>>>(Wk, wkb, (DO * DI) / 4);
  cvt_f32_f16<<<64,  256, 0, stream>>>(Wv, wvb, (DO * DI) / 4);

  proj_f16<<<512, 256, 0, stream>>>(xb, wqb, qh,  0);
  proj_f16<<<512, 256, 0, stream>>>(xb, wkb, kh,  0);
  proj_f16<<<512, 256, 0, stream>>>(xb, wvb, vth, 1);   // V stored transposed

  attn_flash<<<256, 256, 0, stream>>>(qh, kh, vth, out);
}
